// HeteroGNN_17274358464707
// MI455X (gfx1250) — matrix operand order repaired and checked by reading, not compiled, not run on hardware
//
#include <hip/hip_runtime.h>
#include <cstdint>
#include <cstddef>

// ---------------------------------------------------------------------------
// Types for CDNA5 WMMA (gfx1250, wave32)
// ---------------------------------------------------------------------------
typedef __attribute__((ext_vector_type(16))) _Float16 v16h;
typedef __attribute__((ext_vector_type(8)))  float    v8f;

union Frag16 { v16h h; uint4 u4[2]; };

static constexpr int NUSER = 50000;
static constexpr int NITEM = 50000;
static constexpr int FDIM  = 128;
static constexpr int HIDD  = 128;
static constexpr int OUTD  = 64;

// ---------------------------------------------------------------------------
// Monotone float <-> uint encoding for atomicMax on floats (incl. negatives)
// ---------------------------------------------------------------------------
__device__ __forceinline__ unsigned fenc(float f) {
  unsigned u = __float_as_uint(f);
  return (u & 0x80000000u) ? ~u : (u | 0x80000000u);
}
__device__ __forceinline__ float fdec(unsigned e) {
  unsigned u = (e & 0x80000000u) ? (e & 0x7FFFFFFFu) : ~e;
  return __uint_as_float(u);
}
static constexpr unsigned ENC_NEG_INF = 0x007FFFFFu;   // fenc(-inf)

// ---------------------------------------------------------------------------
// Small utility kernels
// ---------------------------------------------------------------------------
__global__ void k_fill_u32(unsigned* p, unsigned v, int n) {
  int i = blockIdx.x * blockDim.x + threadIdx.x;
  if (i < n) p[i] = v;
}
__global__ void k_fill_f32(float* p, float v, int n) {
  int i = blockIdx.x * blockDim.x + threadIdx.x;
  if (i < n) p[i] = v;
}
__global__ void k_cvt_f16(const float* __restrict__ in, unsigned short* __restrict__ out, int n) {
  int i = blockIdx.x * blockDim.x + threadIdx.x;
  if (i < n) {
    union { _Float16 f; unsigned short s; } c;
    c.f = (_Float16)in[i];
    out[i] = c.s;
  }
}
// fused ReLU + cast: out16 = (half)max(0, in); the f32 buffer is not rewritten
__global__ void k_relu_cvt(const float* __restrict__ in, unsigned short* __restrict__ out, int n) {
  int i = blockIdx.x * blockDim.x + threadIdx.x;
  if (i < n) {
    float x = in[i];
    x = x > 0.f ? x : 0.f;
    union { _Float16 f; unsigned short s; } c;
    c.f = (_Float16)x;
    out[i] = c.s;
  }
}
// out[node][c] = bA[c] (+ bB[c] if bB != null); C must be a power of two.
__global__ void k_init_bias(float* __restrict__ out, const float* __restrict__ bA,
                            const float* __restrict__ bB, int n, int C) {
  int i = blockIdx.x * blockDim.x + threadIdx.x;
  if (i < n) {
    int c = i & (C - 1);
    float v = bA[c];
    if (bB) v += bB[c];
    out[i] = v;
  }
}

// ---------------------------------------------------------------------------
// Weight pack: f32 W[K][N] -> B-fragment dwords.
// GEMM reads: Wp[((nt*KT + kt)*32 + lane)*8 + v] holding halves for
// K = kt*32 + (lane>=16 ? 8 : 0) + (v>=4 ? 16 : 0) + 2(v&3), +1 at column N = nt*16 + (lane&15)
// (16-bit B 32x16 layout per CDNA5 ISA 7.12.2 / sparse-B pattern).
// ---------------------------------------------------------------------------
__global__ void k_pack_w(const float* __restrict__ W, unsigned* __restrict__ Wp,
                         int N, int K) {
  int idx = blockIdx.x * blockDim.x + threadIdx.x;
  int KT = K / 32;
  int total = (N / 16) * KT * 32 * 8;
  if (idx >= total) return;
  int v    = idx & 7;
  int lane = (idx >> 3) & 31;
  int t    = idx >> 8;          // nt*KT + kt
  int kt   = t % KT;
  int nt   = t / KT;
  int col  = nt * 16 + (lane & 15);
  int k    = kt * 32 + (lane >> 4) * 8 + (v >> 2) * 16 + 2 * (v & 3);
  union { _Float16 f[2]; unsigned u; } p;
  p.f[0] = (_Float16)W[(size_t)k * N + col];
  p.f[1] = (_Float16)W[(size_t)(k + 1) * N + col];
  Wp[idx] = p.u;
}

// ---------------------------------------------------------------------------
// Fused WMMA GEMM + attention scalar:
//   H[M x N] (f32) = X16[M x K] (f16) * W (packed)       [only if WRITEH]
//   e[i]           = dot(H[i,:], av[:])                   [always]
// One wave computes a 16-row x N strip. K mult of 32, N mult of 16,
// M mult of 16 (50000 = 3125*16). A-fragment per ISA 16-bit 16x32 layout.
// Epilogue: C/D layout puts (row m0+rr+8*hh, col nt*16+r) in acc[nt][rr] of
// lane (hh*16+r), so e is an in-register dot + 4-step shfl_xor over r-bits.
// ---------------------------------------------------------------------------
template <int N, int K, bool WRITEH>
__global__ void __launch_bounds__(128)
k_gemm_fused(const unsigned short* __restrict__ X, const unsigned* __restrict__ Wp,
             float* __restrict__ H, const float* __restrict__ av,
             float* __restrict__ evec, int M) {
  constexpr int KT = K / 32, NT = N / 16;
  int wid  = blockIdx.x * (blockDim.x >> 5) + (threadIdx.x >> 5);
  int lane = threadIdx.x & 31;
  int m0   = wid * 16;
  if (m0 >= M) return;
  int r  = lane & 15;
  int hh = lane >> 4;
  const unsigned* xrow = (const unsigned*)(X + (size_t)(m0 + r) * K);

  v8f acc[NT];
  const v8f vzero = {0.f, 0.f, 0.f, 0.f, 0.f, 0.f, 0.f, 0.f};
#pragma unroll
  for (int nt = 0; nt < NT; nt++) acc[nt] = vzero;

#pragma unroll
  for (int kt = 0; kt < KT; kt++) {
    Frag16 a;
    a.u4[0] = *(const uint4*)(xrow + kt * 16 + hh * 4);
    a.u4[1] = *(const uint4*)(xrow + kt * 16 + 8 + hh * 4);
#pragma unroll
    for (int nt = 0; nt < NT; nt++) {
      Frag16 b;
      const unsigned* wp = Wp + ((size_t)(nt * KT + kt) * 32 + lane) * 8;
      b.u4[0] = *(const uint4*)(wp);
      b.u4[1] = *(const uint4*)(wp + 4);
      acc[nt] = __builtin_amdgcn_wmma_f32_16x16x32_f16(
          false, a.h, false, b.h, (short)0, acc[nt], false, false);
    }
  }

  if constexpr (WRITEH) {
#pragma unroll
    for (int nt = 0; nt < NT; nt++) {
#pragma unroll
      for (int rr = 0; rr < 8; rr++) {
        H[(size_t)(m0 + rr + 8 * hh) * N + nt * 16 + r] = acc[nt][rr];
      }
    }
  }

  // attention scalar: e[m0 + rr + 8*hh] = sum_nt sum_r acc[nt][rr]*av[nt*16+r]
  float al[NT];
#pragma unroll
  for (int nt = 0; nt < NT; nt++) al[nt] = av[nt * 16 + r];
#pragma unroll
  for (int rr = 0; rr < 8; rr++) {
    float p = 0.f;
#pragma unroll
    for (int nt = 0; nt < NT; nt++) p += acc[nt][rr] * al[nt];
    // reduce over the 16 lanes of this half (xor of r-bits keeps hh fixed)
#pragma unroll
    for (int off = 8; off > 0; off >>= 1) p += __shfl_xor(p, off, 32);
    if (r == 0) evec[m0 + rr + 8 * hh] = p;
  }
}

// ---------------------------------------------------------------------------
// Edge pass 1: logit = leaky(e_src[src] + e_dst[dst]); segment-max via
// monotone-encoded atomicMax.
// ---------------------------------------------------------------------------
__global__ void k_edge_logit_max(const int* __restrict__ ei, int E,
                                 const float* __restrict__ es, const float* __restrict__ ed,
                                 float* __restrict__ logit, unsigned* __restrict__ menc) {
  int e = blockIdx.x * blockDim.x + threadIdx.x;
  if (e >= E) return;
  int s = ei[e], d = ei[E + e];
  float x = es[s] + ed[d];
  x = x > 0.f ? x : 0.2f * x;
  logit[e] = x;
  atomicMax(&menc[d], fenc(x));
}

// Edge pass 2: ex = exp(logit - m[dst]); den[dst] += ex
__global__ void k_edge_exp_sum(const int* __restrict__ ei, int E,
                               const float* __restrict__ logit,
                               const unsigned* __restrict__ menc,
                               float* __restrict__ exbuf, float* __restrict__ den) {
  int e = blockIdx.x * blockDim.x + threadIdx.x;
  if (e >= E) return;
  int d = ei[E + e];
  float m  = fdec(menc[d]);           // finite: d has at least this edge
  float ex = expf(logit[e] - m);
  exbuf[e] = ex;
  atomicAdd(&den[d], ex);
}

// Edge pass 3: out[dst][:] += alpha * h_src[src][:] — one wave per edge,
// vectorized b128/b64 gather + per-channel f32 atomic adds.
template <int C>
__global__ void k_edge_aggregate(const int* __restrict__ ei, int E,
                                 const float* __restrict__ exbuf,
                                 const float* __restrict__ den,
                                 const float* __restrict__ hsrc,
                                 float* __restrict__ out) {
  int wid  = blockIdx.x * (blockDim.x >> 5) + (threadIdx.x >> 5);
  int lane = threadIdx.x & 31;
  if (wid >= E) return;
  int s = ei[wid], d = ei[E + wid];
  float alpha = exbuf[wid] / (den[d] + 1e-16f);
  constexpr int PER = C / 32;           // 4 (C=128) or 2 (C=64)
  const float* hs = hsrc + (size_t)s * C + lane * PER;
  float*       op = out  + (size_t)d * C + lane * PER;
  if constexpr (PER == 4) {
    float4 v = *(const float4*)hs;
    atomicAdd(&op[0], alpha * v.x);
    atomicAdd(&op[1], alpha * v.y);
    atomicAdd(&op[2], alpha * v.z);
    atomicAdd(&op[3], alpha * v.w);
  } else {
    float2 v = *(const float2*)hs;
    atomicAdd(&op[0], alpha * v.x);
    atomicAdd(&op[1], alpha * v.y);
  }
}

// ---------------------------------------------------------------------------
// Bilinear edge scoring: p[l] = sum_c osrc[a][c] * relw[c] * odst[b][c]
// ---------------------------------------------------------------------------
template <int C>
__global__ void k_score(const int* __restrict__ eli, int L,
                        const float* __restrict__ osrc, const float* __restrict__ odst,
                        const float* __restrict__ relw, float* __restrict__ out) {
  int wid  = blockIdx.x * (blockDim.x >> 5) + (threadIdx.x >> 5);
  int lane = threadIdx.x & 31;
  if (wid >= L) return;
  int a = eli[wid], b = eli[L + wid];
  constexpr int PER = C / 32;           // 2 for C=64
  static_assert(PER == 2, "score kernel assumes C == 64");
  float2 va = *(const float2*)(osrc + (size_t)a * C + lane * 2);
  float2 vb = *(const float2*)(odst + (size_t)b * C + lane * 2);
  float2 vw = *(const float2*)(relw + lane * 2);
  float s = va.x * vw.x * vb.x + va.y * vw.y * vb.y;
#pragma unroll
  for (int off = 16; off > 0; off >>= 1) s += __shfl_xor(s, off, 32);
  if (lane == 0) out[wid] = s;
}

// ---------------------------------------------------------------------------
// One GAT relation (templated on feature dims)
// ---------------------------------------------------------------------------
template <int K, int N>
static void run_relation(hipStream_t stream,
                         const unsigned short* src16, int Ns,
                         const unsigned short* dst16, int Nd,
                         const float* Ws, const float* Wd,
                         const float* av_s, const float* av_d,
                         const int* ei, int Ecnt,
                         float* hsrc, float* e_src, float* e_dst,
                         unsigned* m_enc, float* den, float* logit, float* exbuf,
                         unsigned* wpack, float* outBuf) {
  int packTotal = (N / 16) * (K / 32) * 32 * 8;
  dim3 blk(256);
  // src projection (materialized) + fused attention scalar
  k_pack_w<<<(packTotal + 255) / 256, blk, 0, stream>>>(Ws, wpack, N, K);
  k_gemm_fused<N, K, true><<<(((Ns + 15) / 16) + 3) / 4, 128, 0, stream>>>(
      src16, wpack, hsrc, av_s, e_src, Ns);
  // dst projection: only the attention scalar is needed -> no H materialization
  k_pack_w<<<(packTotal + 255) / 256, blk, 0, stream>>>(Wd, wpack, N, K);
  k_gemm_fused<N, K, false><<<(((Nd + 15) / 16) + 3) / 4, 128, 0, stream>>>(
      dst16, wpack, nullptr, av_d, e_dst, Nd);
  // segment softmax + aggregate
  k_fill_u32<<<(Nd + 255) / 256, blk, 0, stream>>>(m_enc, ENC_NEG_INF, Nd);
  k_fill_f32<<<(Nd + 255) / 256, blk, 0, stream>>>(den, 0.f, Nd);
  k_edge_logit_max<<<(Ecnt + 255) / 256, blk, 0, stream>>>(ei, Ecnt, e_src, e_dst, logit, m_enc);
  k_edge_exp_sum<<<(Ecnt + 255) / 256, blk, 0, stream>>>(ei, Ecnt, logit, m_enc, exbuf, den);
  k_edge_aggregate<N><<<(Ecnt + 7) / 8, blk, 0, stream>>>(ei, Ecnt, exbuf, den, hsrc, outBuf);
}

// ---------------------------------------------------------------------------
// Orchestration
// ---------------------------------------------------------------------------
extern "C" void kernel_launch(void* const* d_in, const int* in_sizes, int n_in,
                              void* d_out, int out_size, void* d_ws, size_t ws_size,
                              hipStream_t stream) {
  (void)n_in; (void)out_size; (void)ws_size;
  const float* x_user  = (const float*)d_in[0];
  const float* x_item  = (const float*)d_in[1];
  const float* W1_src  = (const float*)d_in[2];   // [3][128][128]
  const float* W1_dst  = (const float*)d_in[3];
  const float* a1_src  = (const float*)d_in[4];   // [3][128]
  const float* a1_dst  = (const float*)d_in[5];
  const float* b1      = (const float*)d_in[6];   // [3][128]
  const float* W2_src  = (const float*)d_in[7];   // [3][128][64]
  const float* W2_dst  = (const float*)d_in[8];
  const float* a2_src  = (const float*)d_in[9];   // [3][64]
  const float* a2_dst  = (const float*)d_in[10];
  const float* b2      = (const float*)d_in[11];  // [3][64]
  const float* rel_w   = (const float*)d_in[12];  // [3][64]
  const int* ei_rates   = (const int*)d_in[13];
  const int* ei_rev     = (const int*)d_in[14];
  const int* ei_follows = (const int*)d_in[15];
  const int* eli_rates   = (const int*)d_in[16];
  const int* eli_rev     = (const int*)d_in[17];
  const int* eli_follows = (const int*)d_in[18];
  float* out = (float*)d_out;

  const int E = in_sizes[13] / 2;    // 500000
  const int L = in_sizes[16] / 2;    // 100000

  // ---- workspace layout (256B aligned) ----
  char* cur = (char*)d_ws;
  auto alloc = [&](size_t bytes) -> void* {
    void* r = (void*)cur;
    cur += (bytes + 255) & ~(size_t)255;
    return r;
  };
  unsigned short* xu16 = (unsigned short*)alloc((size_t)NUSER * FDIM * 2); // reused: h1u16
  unsigned short* xi16 = (unsigned short*)alloc((size_t)NITEM * FDIM * 2); // reused: h1i16
  float* hsrc    = (float*)alloc((size_t)NUSER * HIDD * 4);
  float* h1_user = (float*)alloc((size_t)NUSER * HIDD * 4);  // reused: o_user (f32, OUT<=HID)
  float* h1_item = (float*)alloc((size_t)NITEM * HIDD * 4);  // reused: o_item
  float* e_src   = (float*)alloc((size_t)NUSER * 4);
  float* e_dst   = (float*)alloc((size_t)NUSER * 4);
  unsigned* m_enc = (unsigned*)alloc((size_t)NUSER * 4);
  float* den     = (float*)alloc((size_t)NUSER * 4);
  float* logit   = (float*)alloc((size_t)E * 4);
  float* exbuf   = (float*)alloc((size_t)E * 4);
  unsigned* wpack = (unsigned*)alloc((size_t)(HIDD / 16) * (FDIM / 32) * 32 * 8 * 4);

  dim3 blk(256);

  // ---- cast inputs to f16 ----
  k_cvt_f16<<<((NUSER * FDIM) + 255) / 256, blk, 0, stream>>>(x_user, xu16, NUSER * FDIM);
  k_cvt_f16<<<((NITEM * FDIM) + 255) / 256, blk, 0, stream>>>(x_item, xi16, NITEM * FDIM);

  // ---- layer 1: init outputs with biases (HeteroConv sums biases of incoming rels)
  k_init_bias<<<((NITEM * HIDD) + 255) / 256, blk, 0, stream>>>(h1_item, b1 + 0 * HIDD, nullptr, NITEM * HIDD, HIDD);
  k_init_bias<<<((NUSER * HIDD) + 255) / 256, blk, 0, stream>>>(h1_user, b1 + 1 * HIDD, b1 + 2 * HIDD, NUSER * HIDD, HIDD);

  // rates: user -> item
  run_relation<FDIM, HIDD>(stream, xu16, NUSER, xi16, NITEM,
      W1_src + 0 * FDIM * HIDD, W1_dst + 0 * FDIM * HIDD,
      a1_src + 0 * HIDD, a1_dst + 0 * HIDD,
      ei_rates, E, hsrc, e_src, e_dst, m_enc, den, logit, exbuf, wpack, h1_item);
  // rev: item -> user
  run_relation<FDIM, HIDD>(stream, xi16, NITEM, xu16, NUSER,
      W1_src + 1 * FDIM * HIDD, W1_dst + 1 * FDIM * HIDD,
      a1_src + 1 * HIDD, a1_dst + 1 * HIDD,
      ei_rev, E, hsrc, e_src, e_dst, m_enc, den, logit, exbuf, wpack, h1_user);
  // follows: user -> user
  run_relation<FDIM, HIDD>(stream, xu16, NUSER, xu16, NUSER,
      W1_src + 2 * FDIM * HIDD, W1_dst + 2 * FDIM * HIDD,
      a1_src + 2 * HIDD, a1_dst + 2 * HIDD,
      ei_follows, E, hsrc, e_src, e_dst, m_enc, den, logit, exbuf, wpack, h1_user);

  // fused relu + cast to f16 (into the now-dead x16 regions); the post-ReLU
  // f32 tensor is never consumed, so it is not materialized.
  unsigned short* h1u16 = xu16;
  unsigned short* h1i16 = xi16;
  k_relu_cvt<<<((NUSER * HIDD) + 255) / 256, blk, 0, stream>>>(h1_user, h1u16, NUSER * HIDD);
  k_relu_cvt<<<((NITEM * HIDD) + 255) / 256, blk, 0, stream>>>(h1_item, h1i16, NITEM * HIDD);

  // ---- layer 2 (f32 h1 buffers are dead after the casts -> reuse as outputs)
  float* o_user = h1_user;
  float* o_item = h1_item;
  k_init_bias<<<((NITEM * OUTD) + 255) / 256, blk, 0, stream>>>(o_item, b2 + 0 * OUTD, nullptr, NITEM * OUTD, OUTD);
  k_init_bias<<<((NUSER * OUTD) + 255) / 256, blk, 0, stream>>>(o_user, b2 + 1 * OUTD, b2 + 2 * OUTD, NUSER * OUTD, OUTD);

  run_relation<HIDD, OUTD>(stream, h1u16, NUSER, h1i16, NITEM,
      W2_src + 0 * HIDD * OUTD, W2_dst + 0 * HIDD * OUTD,
      a2_src + 0 * OUTD, a2_dst + 0 * OUTD,
      ei_rates, E, hsrc, e_src, e_dst, m_enc, den, logit, exbuf, wpack, o_item);
  run_relation<HIDD, OUTD>(stream, h1i16, NITEM, h1u16, NUSER,
      W2_src + 1 * HIDD * OUTD, W2_dst + 1 * HIDD * OUTD,
      a2_src + 1 * OUTD, a2_dst + 1 * OUTD,
      ei_rev, E, hsrc, e_src, e_dst, m_enc, den, logit, exbuf, wpack, o_user);
  run_relation<HIDD, OUTD>(stream, h1u16, NUSER, h1u16, NUSER,
      W2_src + 2 * HIDD * OUTD, W2_dst + 2 * HIDD * OUTD,
      a2_src + 2 * OUTD, a2_dst + 2 * OUTD,
      ei_follows, E, hsrc, e_src, e_dst, m_enc, den, logit, exbuf, wpack, o_user);

  // ---- bilinear edge scoring -> d_out [3, L]
  k_score<OUTD><<<(L + 7) / 8, blk, 0, stream>>>(eli_rates,   L, o_user, o_item, rel_w + 0 * OUTD, out + 0 * (size_t)L);
  k_score<OUTD><<<(L + 7) / 8, blk, 0, stream>>>(eli_rev,     L, o_item, o_user, rel_w + 1 * OUTD, out + 1 * (size_t)L);
  k_score<OUTD><<<(L + 7) / 8, blk, 0, stream>>>(eli_follows, L, o_user, o_user, rel_w + 2 * OUTD, out + 2 * (size_t)L);
}
